// DecoderTransformer_80229989089350
// MI455X (gfx1250) — matrix operand order repaired and checked by reading, not compiled, not run on hardware
//
#include <hip/hip_runtime.h>

// ---------------------------------------------------------------------------
// CDNA5 (gfx1250) GPT-2-style decoder forward.
// bf16 WMMA for all matmuls; TDM (tensor_load_to_lds) feeds flash-attention.
// ---------------------------------------------------------------------------

typedef __attribute__((ext_vector_type(16))) __bf16          v16bf;
typedef __attribute__((ext_vector_type(8)))  float           v8f;
typedef __attribute__((ext_vector_type(8)))  unsigned short  us8;
typedef __attribute__((ext_vector_type(4)))  unsigned int    v4u;
typedef __attribute__((ext_vector_type(8)))  int             v8i;
typedef __attribute__((ext_vector_type(4)))  int             v4i;

union Frag { us8 h[2]; v16bf v; };

__device__ __forceinline__ unsigned short f2bf(float f) {
  unsigned u = __float_as_uint(f);
  u += 0x7FFFu + ((u >> 16) & 1u);   // round-to-nearest-even
  return (unsigned short)(u >> 16);
}

__device__ __forceinline__ v8f v8zero() {
  v8f z; for (int i = 0; i < 8; ++i) z[i] = 0.f; return z;
}

__device__ __forceinline__ v8f wmma_bf16(const Frag& a, const Frag& b, v8f c) {
  return __builtin_amdgcn_wmma_f32_16x16x32_bf16(false, a.v, false, b.v,
                                                 (short)0, c, false, false);
}

__device__ __forceinline__ float redmax16(float v) {
  for (int m = 8; m; m >>= 1) v = fmaxf(v, __shfl_xor(v, m, 32));
  return v;
}
__device__ __forceinline__ float redsum16(float v) {
  for (int m = 8; m; m >>= 1) v += __shfl_xor(v, m, 32);
  return v;
}

// ---------------------------------------------------------------------------
// TDM: 2-D tile load (bf16 elements) global -> LDS.
// Tile = tile_d1 rows x tile_d0 elems; row stride in global = stride_elems.
// D# pad fields add 4 DWORDs of LDS padding after every 32 stored DWORDs
// (= one 64-elem bf16 row), yielding an LDS row stride of 72 ushorts.
// ---------------------------------------------------------------------------
__device__ __forceinline__ void tdm_load_2d_bf16(unsigned lds_addr,
                                                 const unsigned short* gptr,
                                                 unsigned tensor_d0, unsigned tensor_d1,
                                                 unsigned tile_d0, unsigned tile_d1,
                                                 unsigned long long stride_elems)
{
  const unsigned long long ga = (unsigned long long)(uintptr_t)gptr;
  v4u g0;
  g0[0] = 1u;                                             // count=1, user mode
  g0[1] = lds_addr;                                       // lds_addr [63:32]
  g0[2] = (unsigned)(ga & 0xFFFFFFFFu);                   // global_addr lo
  g0[3] = (unsigned)((ga >> 32) & 0x01FFFFFFu)            // global_addr hi (57b)
        | 0x80000000u;                                    // type=2 ("image")
  v8i g1;
  g1[0] = (int)((1u << 16)          // data_size = 1 -> 2 bytes
              | (1u << 20)          // pad_enable
              | (4u << 22)          // pad_interval: 32 DWORDs (one 128B row)
              | (3u << 25));        // pad_amount: 4 DWORDs (16B)
  g1[1] = (int)((tensor_d0 & 0xFFFFu) << 16);                              // dim0 lo16 @48
  g1[2] = (int)(((tensor_d0 >> 16) & 0xFFFFu) | ((tensor_d1 & 0xFFFFu) << 16));
  g1[3] = (int)(((tensor_d1 >> 16) & 0xFFFFu) | ((tile_d0 & 0xFFFFu) << 16));
  g1[4] = (int)(tile_d1 & 0xFFFFu);                                        // tile_dim1; tile_dim2=0
  g1[5] = (int)(unsigned)(stride_elems & 0xFFFFFFFFu);                     // dim0_stride lo32
  g1[6] = (int)(unsigned)((stride_elems >> 32) & 0xFFFFu);                 // dim0_stride hi16
  g1[7] = 0;
  const v4i z4 = {0, 0, 0, 0};
#if __clang_major__ >= 23
  const v8i z8 = {0, 0, 0, 0, 0, 0, 0, 0};
  __builtin_amdgcn_tensor_load_to_lds(g0, g1, z4, z4, z8, 0);
#else
  __builtin_amdgcn_tensor_load_to_lds(g0, g1, z4, z4, 0);
#endif
}

// ---------------------------------------------------------------------------
// Generic bf16-WMMA GEMM:  C[M,N] = op(A[M,K] * B[K,N] + bias (+ resid))
//   HEADED: B is [H, K, 64] head-blocked (QKV weights), N = H*64
// Block 256 thr (8 waves, 4x2), tile 128x128, K-step 32.
// ---------------------------------------------------------------------------
#define GA_STR 40   // ushort stride of LDS A rows (32 + 8 pad) -> 80B, 16B-aligned
#define GB_STR 40

template<bool HEADED, bool RELU, bool RESID>
__global__ __launch_bounds__(256)
void gemm_bf16(const float* __restrict__ A, const float* __restrict__ B,
               const float* __restrict__ bias, const float* __restrict__ resid,
               float* __restrict__ C, int M, int N, int K)
{
  __shared__ unsigned short As[128 * GA_STR];
  __shared__ unsigned short Bs[128 * GB_STR];

  const int t    = threadIdx.x;
  const int lane = t & 31;
  const int wave = t >> 5;
  const int wm   = wave & 3;          // 4 waves along M
  const int wn   = wave >> 2;         // 2 waves along N
  const int half = lane >> 4;
  const int lr   = lane & 15;
  const int n0   = blockIdx.x * 128;
  const int m0   = blockIdx.y * 128;

  v8f acc[2][4];
  for (int i = 0; i < 2; ++i) for (int j = 0; j < 4; ++j) acc[i][j] = v8zero();

  for (int k0 = 0; k0 < K; k0 += 32) {
    // ---- stage A tile (128 x 32 fp32 -> bf16, row-major) ----
    {
      const int kq = (t & 7) * 4;
      const int rb = t >> 3;                       // 0..31
      for (int i = 0; i < 4; ++i) {
        const int r = rb + i * 32;
        float4 f = *(const float4*)(A + (size_t)(m0 + r) * K + k0 + kq);
        unsigned short* d = &As[r * GA_STR + kq];
        d[0] = f2bf(f.x); d[1] = f2bf(f.y); d[2] = f2bf(f.z); d[3] = f2bf(f.w);
      }
      if (k0 + 32 < K)
        __builtin_prefetch(A + (size_t)(m0 + rb) * K + k0 + 32 + kq, 0, 3);
    }
    // ---- stage B tile transposed (32 x 128 fp32 -> bf16 [n][k]) ----
    {
      const int nq = (t & 31) * 4;                 // 0..124 (never crosses 64-bdry)
      const int kr = t >> 5;                       // 0..7
      for (int i = 0; i < 4; ++i) {
        const int k = kr + i * 8;
        const int n = n0 + nq;
        float4 f = make_float4(0.f, 0.f, 0.f, 0.f);
        if (HEADED) {
          f = *(const float4*)(B + ((size_t)(n >> 6) * K + (k0 + k)) * 64 + (n & 63));
        } else if (n + 3 < N) {
          f = *(const float4*)(B + (size_t)(k0 + k) * N + n);
        } else {
          const float* p = B + (size_t)(k0 + k) * N;
          if (n     < N) f.x = p[n];
          if (n + 1 < N) f.y = p[n + 1];
          if (n + 2 < N) f.z = p[n + 2];
        }
        Bs[(nq + 0) * GB_STR + k] = f2bf(f.x);
        Bs[(nq + 1) * GB_STR + k] = f2bf(f.y);
        Bs[(nq + 2) * GB_STR + k] = f2bf(f.z);
        Bs[(nq + 3) * GB_STR + k] = f2bf(f.w);
      }
    }
    __syncthreads();

    // ---- fragments + 8 WMMA per wave ----
    Frag af[2], bf[4];
    for (int mi = 0; mi < 2; ++mi) {
      const int r = wm * 32 + mi * 16 + lr;
      af[mi].h[0] = *(const us8*)&As[r * GA_STR + half * 8];
      af[mi].h[1] = *(const us8*)&As[r * GA_STR + 16 + half * 8];
    }
    for (int ni = 0; ni < 4; ++ni) {
      const int n = wn * 64 + ni * 16 + lr;
      const unsigned short* p = &Bs[n * GB_STR + half * 8];
      bf[ni].h[0] = *(const us8*)(p);
      bf[ni].h[1] = *(const us8*)(p + 16);
    }
    for (int mi = 0; mi < 2; ++mi)
      for (int ni = 0; ni < 4; ++ni)
        acc[mi][ni] = wmma_bf16(af[mi], bf[ni], acc[mi][ni]);

    __syncthreads();
  }

  // ---- epilogue ----
  for (int mi = 0; mi < 2; ++mi)
    for (int ni = 0; ni < 4; ++ni) {
      const int col = n0 + wn * 64 + ni * 16 + lr;
      if (col >= N) continue;
      const float bv = bias ? bias[col] : 0.f;
      for (int v = 0; v < 8; ++v) {
        const int row = m0 + wm * 32 + mi * 16 + half * 8 + v;
        float x = acc[mi][ni][v] + bv;
        if (RELU)  x = fmaxf(x, 0.f);
        if (RESID) x += resid[(size_t)row * N + col];
        C[(size_t)row * N + col] = x;
      }
    }
}

// ---------------------------------------------------------------------------
// K/V repack to bf16 for the TDM path:
//   kbf[b,h,t,d]  (row-major, 64 contiguous d)    from kb[b,t,h*64+d]
//   vbf[b,h,d,t]  (transposed, 2048 contiguous t) from vb[b,t,h*64+d]
// ---------------------------------------------------------------------------
__global__ __launch_bounds__(256)
void kv_bf16_k(const float* __restrict__ kb, const float* __restrict__ vb,
               unsigned short* __restrict__ kbf, unsigned short* __restrict__ vbf)
{
  const int row = blockIdx.x;              // b*2048 + t
  const int b   = row >> 11;
  const int tt  = row & 2047;
  for (int j = 0; j < 2; ++j) {
    const int cc = threadIdx.x * 2 + j;
    const int hh = cc >> 6, d = cc & 63;
    const int bh = b * 8 + hh;
    kbf[((size_t)bh * 2048 + tt) * 64 + d] = f2bf(kb[(size_t)row * 512 + cc]);
    vbf[((size_t)bh * 64 + d) * 2048 + tt] = f2bf(vb[(size_t)row * 512 + cc]);
  }
}

// ---------------------------------------------------------------------------
// Flash attention (causal, HD=64, T=2048, H=8, B=2), TDM-fed K/V tiles.
// Block = 256 thr (8 waves); wave owns 32 query rows; 64-key tiles,
// double-buffered in LDS; wave 0 drives the Tensor Data Mover.
// Dynamic LDS layout (ushorts):
//   [0      .. 4608)   Ks buf0   [key][d]  stride 72 (TDM-padded)
//   [4608   .. 9216)   Ks buf1
//   [9216   .. 13824)  Vs buf0   [d][key]  stride 72
//   [13824  .. 18432)  Vs buf1
//   [18432  .. 36864)  Ps: 8 waves x [32 rows][72]
// ---------------------------------------------------------------------------
#define ATT_STR 72

__global__ __launch_bounds__(256)
void attn_k(const float* __restrict__ Qg, const unsigned short* __restrict__ Kbf,
            const unsigned short* __restrict__ Vbf, float* __restrict__ O)
{
  extern __shared__ unsigned short smem[];

  const int t    = threadIdx.x;
  const int lane = t & 31;
  const int wave = t >> 5;
  const int half = lane >> 4;
  const int lr   = lane & 15;
  const int bh   = blockIdx.y;             // b*8 + h
  const int b    = bh >> 3;
  const int hh   = bh & 7;
  const int qblock = blockIdx.x * 256;
  const int qbase  = qblock + wave * 32;

  const float* Qb = Qg + (size_t)b * 2048 * 512 + hh * 64;
  const unsigned short* Kh = Kbf + (size_t)bh * 2048 * 64;   // [t][d]
  const unsigned short* Vh = Vbf + (size_t)bh * 64 * 2048;   // [d][t]

  // ---- load Q fragments (2 q-subtiles x 2 k-steps over HD=64) ----
  Frag qf[2][2];
  for (int qi = 0; qi < 2; ++qi) {
    const float* qrow = Qb + (size_t)(qbase + qi * 16 + lr) * 512;
    for (int kd = 0; kd < 2; ++kd) {
      unsigned short tmp[16];
      for (int j = 0; j < 8; ++j) tmp[j]     = f2bf(qrow[kd * 32 + half * 8 + j]);
      for (int j = 0; j < 8; ++j) tmp[8 + j] = f2bf(qrow[kd * 32 + 16 + half * 8 + j]);
      __builtin_memcpy(&qf[qi][kd], tmp, 32);
    }
  }

  v8f oacc[2][4];
  for (int i = 0; i < 2; ++i) for (int j = 0; j < 4; ++j) oacc[i][j] = v8zero();
  float mrow[2][8], lrow[2][8];
  for (int qi = 0; qi < 2; ++qi)
    for (int v = 0; v < 8; ++v) { mrow[qi][v] = -3.0e38f; lrow[qi][v] = 0.f; }

  const int ktmax_w = (qbase + 31) >> 6;
  const int ktmax_b = (qblock + 255) >> 6;
  const float sc = 0.125f;                       // 1/sqrt(64)

  const unsigned lds0 = (unsigned)(uintptr_t)smem;   // LDS byte address of smem

  // ---- prologue: DMA tile 0 into buffer 0 (wave 0 only; TDM ignores EXEC) ----
  if (wave == 0) {
    tdm_load_2d_bf16(lds0 +  0 * 2,        Kh, 64, 2048, 64, 64, 64);
    tdm_load_2d_bf16(lds0 + 9216 * 2 / 2 * 2, Vh, 2048, 64, 64, 64, 2048);
  }

  for (int kt = 0; kt <= ktmax_b; ++kt) {
    if (wave == 0) __builtin_amdgcn_s_wait_tensorcnt(0);   // tile kt landed
    __syncthreads();

    // ---- kick off DMA for tile kt+1 into the other buffer (overlaps WMMA) ----
    const int buf = kt & 1;
    if (wave == 0 && kt < ktmax_b) {
      const int nb = (kt + 1) & 1;
      const unsigned kN = (unsigned)(kt + 1) * 64u;
      tdm_load_2d_bf16(lds0 + (unsigned)(nb * 4608) * 2,
                       Kh + (size_t)kN * 64, 64, 2048 - kN, 64, 64, 64);
      tdm_load_2d_bf16(lds0 + (unsigned)(9216 + nb * 4608) * 2,
                       Vh + kN, 2048 - kN, 64, 64, 64, 2048);
    }

    if (kt <= ktmax_w) {
      const unsigned short* Ks = smem + buf * 4608;
      const unsigned short* Vs = smem + 9216 + buf * 4608;
      unsigned short*       Ps = smem + 18432 + wave * (32 * ATT_STR);

      // ---- S = Q K^T, mask, online softmax, emit P (bf16) into LDS ----
      for (int qi = 0; qi < 2; ++qi) {
        v8f s[4];
        for (int kj = 0; kj < 4; ++kj) {
          const unsigned short* kp = &Ks[(kj * 16 + lr) * ATT_STR];
          Frag bfr;
          v8f a = v8zero();
          bfr.h[0] = *(const us8*)(kp + half * 8);
          bfr.h[1] = *(const us8*)(kp + 16 + half * 8);
          a = wmma_bf16(qf[qi][0], bfr, a);
          bfr.h[0] = *(const us8*)(kp + 32 + half * 8);
          bfr.h[1] = *(const us8*)(kp + 32 + 16 + half * 8);
          a = wmma_bf16(qf[qi][1], bfr, a);
          s[kj] = a;
        }
        for (int v = 0; v < 8; ++v) {
          const int qidx = qbase + qi * 16 + half * 8 + v;
          float mx = -3.0e38f;
          for (int kj = 0; kj < 4; ++kj) {
            const int key = kt * 64 + kj * 16 + lr;
            float x = s[kj][v] * sc;
            if (key > qidx) x = -3.0e38f;
            s[kj][v] = x;
            mx = fmaxf(mx, x);
          }
          mx = redmax16(mx);
          const float mnew  = fmaxf(mrow[qi][v], mx);
          const float alpha = __expf(mrow[qi][v] - mnew);
          mrow[qi][v] = mnew;
          lrow[qi][v] *= alpha;
          for (int n = 0; n < 4; ++n) oacc[qi][n][v] *= alpha;
          float ps = 0.f;
          for (int kj = 0; kj < 4; ++kj) {
            const float p = __expf(s[kj][v] - mnew);
            ps += p;
            Ps[(qi * 16 + half * 8 + v) * ATT_STR + kj * 16 + lr] = f2bf(p);
          }
          lrow[qi][v] += redsum16(ps);
        }
      }
      // ---- O += P V ----
      for (int qi = 0; qi < 2; ++qi) {
        Frag pa[2];
        const unsigned short* pr = &Ps[(qi * 16 + lr) * ATT_STR];
        for (int ks = 0; ks < 2; ++ks) {
          pa[ks].h[0] = *(const us8*)(pr + ks * 32 + half * 8);
          pa[ks].h[1] = *(const us8*)(pr + ks * 32 + 16 + half * 8);
        }
        for (int n = 0; n < 4; ++n) {
          const unsigned short* vp = &Vs[(n * 16 + lr) * ATT_STR];
          Frag bfr;
          bfr.h[0] = *(const us8*)(vp + half * 8);
          bfr.h[1] = *(const us8*)(vp + 16 + half * 8);
          oacc[qi][n] = wmma_bf16(pa[0], bfr, oacc[qi][n]);
          bfr.h[0] = *(const us8*)(vp + 32 + half * 8);
          bfr.h[1] = *(const us8*)(vp + 32 + 16 + half * 8);
          oacc[qi][n] = wmma_bf16(pa[1], bfr, oacc[qi][n]);
        }
      }
    }
    __syncthreads();   // all waves done with buf before TDM reuses it
  }

  // ---- write O[b, q, hh*64 + d] = acc / l ----
  for (int qi = 0; qi < 2; ++qi)
    for (int n = 0; n < 4; ++n) {
      const int d = n * 16 + lr;
      for (int v = 0; v < 8; ++v) {
        const int qidx = qbase + qi * 16 + half * 8 + v;
        O[((size_t)b * 2048 + qidx) * 512 + hh * 64 + d] =
            oacc[qi][n][v] / lrow[qi][v];
      }
    }
}

// ---------------------------------------------------------------------------
// LayerNorm over C=512, one row per block (256 thr, 2 elems each).
// ---------------------------------------------------------------------------
__global__ __launch_bounds__(256)
void layernorm_k(const float* __restrict__ x, const float* __restrict__ sc,
                 const float* __restrict__ bi, float* __restrict__ y)
{
  __shared__ float red[256];
  const int row = blockIdx.x, t = threadIdx.x;
  const float* xr = x + (size_t)row * 512;
  float2 v = *(const float2*)(xr + t * 2);
  red[t] = v.x + v.y;
  __syncthreads();
  for (int o = 128; o; o >>= 1) { if (t < o) red[t] += red[t + o]; __syncthreads(); }
  const float mean = red[0] * (1.f / 512.f);
  __syncthreads();
  const float d0 = v.x - mean, d1 = v.y - mean;
  red[t] = d0 * d0 + d1 * d1;
  __syncthreads();
  for (int o = 128; o; o >>= 1) { if (t < o) red[t] += red[t + o]; __syncthreads(); }
  const float rstd = rsqrtf(red[0] * (1.f / 512.f) + 1e-5f);
  const int c = t * 2;
  float2 o2;
  o2.x = d0 * rstd * sc[c]     + bi[c];
  o2.y = d1 * rstd * sc[c + 1] + bi[c + 1];
  *(float2*)(y + (size_t)row * 512 + c) = o2;
}

// ---------------------------------------------------------------------------
__global__ __launch_bounds__(256)
void embed_k(const int* __restrict__ x, const float* __restrict__ te,
             const float* __restrict__ pe, float* __restrict__ h)
{
  const int row = blockIdx.x, t = threadIdx.x;
  const int tok = x[row];
  const int tp  = row & 2047;
  const int c   = t * 2;
  float2 a = *(const float2*)(te + (size_t)tok * 512 + c);
  float2 p = *(const float2*)(pe + (size_t)tp  * 512 + c);
  a.x += p.x; a.y += p.y;
  *(float2*)(h + (size_t)row * 512 + c) = a;
}

// ---------------------------------------------------------------------------
__global__ __launch_bounds__(256)
void rowloss_k(const float* __restrict__ logits, const int* __restrict__ tgt,
               float* __restrict__ rl)
{
  __shared__ float red[256];
  const int row = blockIdx.x, t = threadIdx.x;
  const float* lp = logits + (size_t)row * 50257;
  float mx = -3.0e38f;
  for (int j = t; j < 50257; j += 256) mx = fmaxf(mx, lp[j]);
  red[t] = mx;
  __syncthreads();
  for (int o = 128; o; o >>= 1) { if (t < o) red[t] = fmaxf(red[t], red[t + o]); __syncthreads(); }
  mx = red[0];
  __syncthreads();
  float s = 0.f;
  for (int j = t; j < 50257; j += 256) s += __expf(lp[j] - mx);
  red[t] = s;
  __syncthreads();
  for (int o = 128; o; o >>= 1) { if (t < o) red[t] += red[t + o]; __syncthreads(); }
  if (t == 0) rl[row] = mx + __logf(red[0]) - lp[tgt[row]];
}

__global__ __launch_bounds__(256)
void loss_reduce_k(const float* __restrict__ rl, float* __restrict__ out)
{
  __shared__ float red[256];
  const int t = threadIdx.x;
  float s = 0.f;
  for (int j = t; j < 4096; j += 256) s += rl[j];
  red[t] = s;
  __syncthreads();
  for (int o = 128; o; o >>= 1) { if (t < o) red[t] += red[t + o]; __syncthreads(); }
  if (t == 0) *out = red[0] * (1.f / 4096.f);
}

// ---------------------------------------------------------------------------
extern "C" void kernel_launch(void* const* d_in, const int* in_sizes, int n_in,
                              void* d_out, int out_size, void* d_ws, size_t ws_size,
                              hipStream_t stream)
{
  const int*   x    = (const int*)  d_in[0];
  const int*   tgt  = (const int*)  d_in[1];
  const float* te   = (const float*)d_in[2];
  const float* pe   = (const float*)d_in[3];
  const float* ln1s = (const float*)d_in[4];
  const float* ln1b = (const float*)d_in[5];
  const float* wq   = (const float*)d_in[6];
  const float* wk   = (const float*)d_in[7];
  const float* wv   = (const float*)d_in[8];
  const float* wpj  = (const float*)d_in[9];
  const float* bpj  = (const float*)d_in[10];
  const float* ln2s = (const float*)d_in[11];
  const float* ln2b = (const float*)d_in[12];
  const float* wf1  = (const float*)d_in[13];
  const float* bf1  = (const float*)d_in[14];
  const float* wf2  = (const float*)d_in[15];
  const float* bf2  = (const float*)d_in[16];
  const float* lnfs = (const float*)d_in[17];
  const float* lnfb = (const float*)d_in[18];
  const float* lmw  = (const float*)d_in[19];
  const float* lmb  = (const float*)d_in[20];
  float* logits = (float*)d_out;

  const size_t NT = 4096, C = 512, HID = 2048;
  float* ws  = (float*)d_ws;
  float* h   = ws;                 // [4096,512]
  float* a   = h   + NT * C;       // [4096,512]
  float* q   = a   + NT * C;       // [4096,512]
  float* kb  = q   + NT * C;       // [4096,512]
  float* vb  = kb  + NT * C;       // [4096,512]
  float* atn = vb  + NT * C;       // [4096,512]
  float* mid = atn + NT * C;       // [4096,2048]
  float* rl  = mid + NT * HID;     // [4096]
  unsigned short* kbf = (unsigned short*)(rl + 4096);   // [2,8,2048,64] bf16
  unsigned short* vbf = kbf + NT * C;                   // [2,8,64,2048] bf16

  const dim3 blk(256);
  const dim3 g512(4, 32);          // N=512 tiles x M=4096/128
  const dim3 g2048(16, 32);
  const dim3 gV(393, 32);          // ceil(50257/128)
  const unsigned attn_lds = 36864u * 2u;   // 73,728 B dynamic LDS

  embed_k<<<4096, blk, 0, stream>>>(x, te, pe, h);

  for (int l = 0; l < 4; ++l) {
    const size_t wo = (size_t)l * 512 * 512;
    layernorm_k<<<4096, blk, 0, stream>>>(h, ln1s + l * 512, ln1b + l * 512, a);
    gemm_bf16<true,  false, false><<<g512, blk, 0, stream>>>(a, wq + wo, nullptr, nullptr, q,  4096, 512, 512);
    gemm_bf16<true,  false, false><<<g512, blk, 0, stream>>>(a, wk + wo, nullptr, nullptr, kb, 4096, 512, 512);
    gemm_bf16<true,  false, false><<<g512, blk, 0, stream>>>(a, wv + wo, nullptr, nullptr, vb, 4096, 512, 512);
    kv_bf16_k<<<4096, blk, 0, stream>>>(kb, vb, kbf, vbf);
    attn_k<<<dim3(8, 16), blk, attn_lds, stream>>>(q, kbf, vbf, atn);
    gemm_bf16<false, false, true ><<<g512, blk, 0, stream>>>(atn, wpj + wo, bpj + l * 512, h, h, 4096, 512, 512);
    layernorm_k<<<4096, blk, 0, stream>>>(h, ln2s + l * 512, ln2b + l * 512, a);
    gemm_bf16<false, true,  false><<<g2048, blk, 0, stream>>>(a, wf1 + (size_t)l * 512 * 2048, bf1 + l * 2048, nullptr, mid, 4096, 2048, 512);
    gemm_bf16<false, false, true ><<<g512, blk, 0, stream>>>(mid, wf2 + (size_t)l * 2048 * 512, bf2 + l * 512, h, h, 4096, 512, 2048);
  }

  layernorm_k<<<4096, blk, 0, stream>>>(h, lnfs, lnfb, a);
  gemm_bf16<false, false, false><<<gV, blk, 0, stream>>>(a, lmw, lmb, nullptr, logits, 4096, 50257, 512);
  rowloss_k<<<4096, blk, 0, stream>>>(logits, tgt, rl);
  loss_reduce_k<<<1, blk, 0, stream>>>(rl, logits + (size_t)4096 * 50257);
}
